// MyModel_7378753814826
// MI455X (gfx1250) — matrix operand order repaired and checked by reading, not compiled, not run on hardware
//
#include <hip/hip_runtime.h>
#include <hip/hip_bf16.h>
#include <cstdint>

typedef __attribute__((ext_vector_type(16))) _Float16 v16h;
typedef __attribute__((ext_vector_type(8)))  float    v8f;

#define T_TRIP 10
#define HID    1024
#define K1PAD  320        // 300 padded to multiple of 32
#define MB     64         // rows per block
#define XS     328        // LDS row stride for X (halves), 16B-aligned rows
#define HS     1032       // LDS row stride for H1 (halves), 16B-aligned rows
#define NTHREADS 512      // 16 wave32

#define WMMA16(A, Bv, C) \
  __builtin_amdgcn_wmma_f32_16x16x32_f16(false, (A), false, (Bv), (short)0, (C), false, false)

// ---------------------------------------------------------------------------
// Pack a row-major f32 weight [Ksrc x N] into WMMA B-fragment order (f16):
//   out[((kt*N + n)*2 + h)*16 + i] = W[kt*32 + h*16 + i][n]   (0 if k >= Ksrc)
// so a lane's whole 16-half B fragment is one contiguous 32-byte load.
// ---------------------------------------------------------------------------
__global__ __launch_bounds__(256) void pack_b_f16(const float* __restrict__ W,
                                                  _Float16* __restrict__ out,
                                                  int Kpad, int Ksrc, int N) {
  int total = Kpad * N;
  for (int idx = blockIdx.x * 256 + threadIdx.x; idx < total;
       idx += gridDim.x * 256) {
    int per = 32 * N;
    int kt  = idx / per;
    int rem = idx - kt * per;
    int n   = rem >> 5;
    int h   = (rem >> 4) & 1;
    int i   = rem & 15;
    int k   = kt * 32 + h * 16 + i;
    float v = (k < Ksrc) ? W[(size_t)k * N + n] : 0.0f;
    out[idx] = (_Float16)v;
  }
}

// ---------------------------------------------------------------------------
// Fused: featurize -> GEMM1(relu) -> GEMM2(relu) -> dot(W3)+b3
// Block: 512 threads (16 wave32), 64 structures. Grid: B/64.
// Wave (strip, ch): rows strip*16..+15, cols ch*256..+255.
// ---------------------------------------------------------------------------
__global__ __launch_bounds__(NTHREADS) void fused_mlp(
    const float* __restrict__ S,    // [B,10,3]
    const float* __restrict__ E,    // [118,10]
    const float* __restrict__ b1,   // [1024]
    const float* __restrict__ b2,   // [1024]
    const float* __restrict__ W3,   // [1024]
    const float* __restrict__ b3,   // [1]
    const _Float16* __restrict__ W1p,  // packed [320 x 1024]
    const _Float16* __restrict__ W2p,  // packed [1024 x 1024]
    float* __restrict__ out) {

  __shared__ __align__(16) _Float16 Xs[MB * XS];
  __shared__ __align__(16) _Float16 H1s[MB * HS];
  __shared__ float outP[MB];

  const int tid  = threadIdx.x;
  const int lane = tid & 31;
  const int wave = tid >> 5;
  const int m0   = blockIdx.x * MB;

  if (tid < MB) outP[tid] = 0.0f;

  // ---------------- Stage 1: featurize into LDS (f16) ----------------
  for (int task = tid; task < MB * T_TRIP; task += NTHREADS) {
    int row = task / T_TRIP;
    int t   = task - row * T_TRIP;
    const float* sp = S + ((size_t)(m0 + row) * T_TRIP + t) * 3;
    int   ii = (int)sp[0];
    int   jj = (int)sp[1];
    float d  = sp[2];
    _Float16* xr = &Xs[row * XS + t * 30];
    if (ii != 0) {
      const float* er = E + (size_t)ii * 10;
      #pragma unroll
      for (int c = 0; c < 10; ++c) xr[c] = (_Float16)er[c];
      #pragma unroll
      for (int c = 0; c < 10; ++c) {
        float u = 0.7f * (float)(c + 1) - d;
        xr[20 + c] = (_Float16)__expf(-u * u);
      }
    } else {
      #pragma unroll
      for (int c = 0; c < 10; ++c) { xr[c] = (_Float16)0.0f; xr[20 + c] = (_Float16)0.0f; }
    }
    if (jj != 0) {
      const float* er = E + (size_t)jj * 10;
      #pragma unroll
      for (int c = 0; c < 10; ++c) xr[10 + c] = (_Float16)er[c];
    } else {
      #pragma unroll
      for (int c = 0; c < 10; ++c) xr[10 + c] = (_Float16)0.0f;
    }
  }
  // zero K padding 300..319
  for (int p = tid; p < MB * 20; p += NTHREADS) {
    int row = p / 20;
    Xs[row * XS + 300 + (p - row * 20)] = (_Float16)0.0f;
  }
  __syncthreads();

  // wave -> (row strip, column quarter)
  const int strip  = wave >> 2;              // 0..3, rows strip*16..+15
  const int chbase = (wave & 3) * 256;       // cols chbase..+255
  const int lr     = lane & 15;              // fragment row/col within tile
  const int hsel   = lane >> 4;              // 0/1 half select
  const int arow   = strip * 16 + lr;        // A-fragment source row in LDS
  const int drow   = strip * 16 + hsel * 8;  // D-fragment row base

  // ---------------- Stage 2: H1 = relu(X @ W1 + b1) ----------------
  // 4 groups of 4 column tiles; one A fragment feeds 4 WMMAs.
  #pragma unroll 1
  for (int ctg = 0; ctg < 4; ++ctg) {
    const int col0 = chbase + ctg * 64 + lr;
    v8f acc[4];
    #pragma unroll
    for (int sub = 0; sub < 4; ++sub) {
      const float bias = b1[col0 + sub * 16];
      #pragma unroll
      for (int v = 0; v < 8; ++v) acc[sub][v] = bias;
    }

    #pragma unroll 1
    for (int kt = 0; kt < K1PAD / 32; ++kt) {
      const int kb = kt * 32 + hsel * 16;
      v16h a;
      *((uint4*)&a)     = *(const uint4*)&Xs[arow * XS + kb];
      *((uint4*)&a + 1) = *(const uint4*)&Xs[arow * XS + kb + 8];
      const _Float16* bp = W1p + (((size_t)kt * HID + col0) * 2 + hsel) * 16;
      v16h bf0 = *(const v16h*)(bp);
      v16h bf1 = *(const v16h*)(bp + 512);
      v16h bf2 = *(const v16h*)(bp + 1024);
      v16h bf3 = *(const v16h*)(bp + 1536);
      acc[0] = WMMA16(a, bf0, acc[0]);
      acc[1] = WMMA16(a, bf1, acc[1]);
      acc[2] = WMMA16(a, bf2, acc[2]);
      acc[3] = WMMA16(a, bf3, acc[3]);
    }

    #pragma unroll
    for (int sub = 0; sub < 4; ++sub) {
      const int col = col0 + sub * 16;
      #pragma unroll
      for (int v = 0; v < 8; ++v) {
        float x = acc[sub][v] > 0.0f ? acc[sub][v] : 0.0f;
        H1s[(drow + v) * HS + col] = (_Float16)x;
      }
    }
  }
  __syncthreads();

  // -------- Stage 3: out += relu(H1 @ W2 + b2) . W3  (fused) --------
  float racc[8];
  #pragma unroll
  for (int v = 0; v < 8; ++v) racc[v] = 0.0f;

  #pragma unroll 1
  for (int ctg = 0; ctg < 4; ++ctg) {
    const int col0 = chbase + ctg * 64 + lr;
    v8f acc[4];
    #pragma unroll
    for (int sub = 0; sub < 4; ++sub) {
      const float bias = b2[col0 + sub * 16];
      #pragma unroll
      for (int v = 0; v < 8; ++v) acc[sub][v] = bias;
    }

    #pragma unroll 2
    for (int kt = 0; kt < HID / 32; ++kt) {
      const int kb = kt * 32 + hsel * 16;
      v16h a;
      *((uint4*)&a)     = *(const uint4*)&H1s[arow * HS + kb];
      *((uint4*)&a + 1) = *(const uint4*)&H1s[arow * HS + kb + 8];
      const _Float16* bp = W2p + (((size_t)kt * HID + col0) * 2 + hsel) * 16;
      v16h bf0 = *(const v16h*)(bp);
      v16h bf1 = *(const v16h*)(bp + 512);
      v16h bf2 = *(const v16h*)(bp + 1024);
      v16h bf3 = *(const v16h*)(bp + 1536);
      acc[0] = WMMA16(a, bf0, acc[0]);
      acc[1] = WMMA16(a, bf1, acc[1]);
      acc[2] = WMMA16(a, bf2, acc[2]);
      acc[3] = WMMA16(a, bf3, acc[3]);
    }

    #pragma unroll
    for (int sub = 0; sub < 4; ++sub) {
      const int col = col0 + sub * 16;
      const float w3v = W3[col];
      #pragma unroll
      for (int v = 0; v < 8; ++v) {
        float x = acc[sub][v] > 0.0f ? acc[sub][v] : 0.0f;
        racc[v] += x * w3v;
      }
    }
  }

  // reduce each of the 8 row-partials across the 16 lanes of each half-wave
  #pragma unroll
  for (int v = 0; v < 8; ++v) {
    float s = racc[v];
    s += __shfl_xor(s, 1, 16);
    s += __shfl_xor(s, 2, 16);
    s += __shfl_xor(s, 4, 16);
    s += __shfl_xor(s, 8, 16);
    if (lr == 0) atomicAdd(&outP[drow + v], s);   // ds_add_f32
  }
  __syncthreads();

  if (tid < MB) out[m0 + tid] = outP[tid] + b3[0];
}

// ---------------------------------------------------------------------------
extern "C" void kernel_launch(void* const* d_in, const int* in_sizes, int n_in,
                              void* d_out, int out_size, void* d_ws, size_t ws_size,
                              hipStream_t stream) {
  const float* S  = (const float*)d_in[0];   // [B,10,3]
  const float* E  = (const float*)d_in[1];   // [118,10]
  const float* W1 = (const float*)d_in[2];   // [300,1024]
  const float* b1 = (const float*)d_in[3];   // [1024]
  const float* W2 = (const float*)d_in[4];   // [1024,1024]
  const float* b2 = (const float*)d_in[5];   // [1024]
  const float* W3 = (const float*)d_in[6];   // [1024,1]
  const float* b3 = (const float*)d_in[7];   // [1]
  float* out = (float*)d_out;

  const int B = in_sizes[0] / (T_TRIP * 3);

  _Float16* W1p = (_Float16*)d_ws;                                   // 320*1024 f16
  _Float16* W2p = (_Float16*)((char*)d_ws + (size_t)K1PAD * HID * 2);// 1024*1024 f16

  {
    int n1 = K1PAD * HID;
    pack_b_f16<<<(n1 + 255) / 256, 256, 0, stream>>>(W1, W1p, K1PAD, 300, HID);
    int n2 = HID * HID;
    pack_b_f16<<<(n2 + 255) / 256, 256, 0, stream>>>(W2, W2p, HID, HID, HID);
  }

  fused_mlp<<<B / MB, NTHREADS, 0, stream>>>(S, E, b1, b2, W3, b3, W1p, W2p, out);
}
